// PointTransformerEncoder_46119358825124
// MI455X (gfx1250) — matrix operand order repaired and checked by reading, not compiled, not run on hardware
//
#include <hip/hip_runtime.h>

#define BB   4
#define NN   8192
#define BN   (BB * NN)
#define HID  128
#define LAT  256
#define KNB  16

typedef __attribute__((ext_vector_type(16))) __bf16 v16bf;
typedef __attribute__((ext_vector_type(8)))  __bf16 v8bf;
typedef __attribute__((ext_vector_type(8)))  float  v8f;

union BfFrag { v16bf v; struct { v8bf lo, hi; } p; };

__device__ __forceinline__ __bf16 f2bf(float x) { return (__bf16)x; }

// -------- transpose x(B,3,N) -> pos(B*N,3) --------
__global__ void prep_pos(const float* __restrict__ x, float* __restrict__ pos) {
  int idx = blockIdx.x * blockDim.x + threadIdx.x;
  if (idx >= BN) return;
  int b = idx / NN, n = idx % NN;
  const float* xb = x + (size_t)b * 3 * NN;
  pos[idx * 3 + 0] = xb[n];
  pos[idx * 3 + 1] = xb[NN + n];
  pos[idx * 3 + 2] = xb[2 * NN + n];
}

// -------- brute-force KNN, top-16 (excl self), nbr[.,0]=self --------
__global__ void knn_kernel(const float* __restrict__ pos, int* __restrict__ nbr) {
  int b = blockIdx.y;
  int i = blockIdx.x * blockDim.x + threadIdx.x;  // 0..NN-1
  const float* posb = pos + (size_t)b * NN * 3;
  float px = posb[i * 3 + 0], py = posb[i * 3 + 1], pz = posb[i * 3 + 2];
  float bestd[KNB]; int besti[KNB];
#pragma unroll
  for (int s = 0; s < KNB; s++) { bestd[s] = 3.0e38f; besti[s] = 0; }
  __shared__ float sp[256 * 3];
  for (int tile = 0; tile < NN; tile += 256) {
    for (int t = threadIdx.x; t < 256 * 3; t += blockDim.x)
      sp[t] = posb[(size_t)tile * 3 + t];
    __syncthreads();
    for (int jj = 0; jj < 256; jj++) {
      int j = tile + jj;
      float dx = px - sp[jj * 3 + 0];
      float dy = py - sp[jj * 3 + 1];
      float dz = pz - sp[jj * 3 + 2];
      float d = dx * dx + dy * dy + dz * dz;
      if (j != i && d < bestd[KNB - 1]) {
        float dd = d; int ji = j;
#pragma unroll
        for (int s = 0; s < KNB; s++) {
          bool lt = dd < bestd[s];
          float td = bestd[s]; int ti = besti[s];
          bestd[s] = lt ? dd : td;  besti[s] = lt ? ji : ti;
          dd = lt ? td : dd;        ji = lt ? ti : ji;
        }
      }
    }
    __syncthreads();
  }
  int* out = nbr + ((size_t)b * NN + i) * (KNB + 1);
  out[0] = i;
#pragma unroll
  for (int s = 0; s < KNB; s++) out[1 + s] = besti[s];
}

// -------- layer-1 MLP (K=3, scalar) -> relu -> bf16 --------
__global__ void mlp1_kernel(const float* __restrict__ x, const float* __restrict__ w1,
                            const float* __restrict__ b1, __bf16* __restrict__ t) {
  int idx = blockIdx.x * blockDim.x + threadIdx.x;   // i*HID + c, exact grid
  int i = idx >> 7, c = idx & (HID - 1);
  int b = i / NN, n = i % NN;
  const float* xb = x + (size_t)b * 3 * NN;
  float x0 = xb[n], x1 = xb[NN + n], x2 = xb[2 * NN + n];
  float v = fmaf(x0, w1[c], fmaf(x1, w1[HID + c], fmaf(x2, w1[2 * HID + c], b1[c])));
  t[idx] = f2bf(fmaxf(v, 0.0f));
}

// -------- weight (K,N) f32 -> (N,K) bf16 transpose --------
__global__ void wcvt_kernel(const float* __restrict__ w, __bf16* __restrict__ wt,
                            int K, int Nn) {
  int idx = blockIdx.x * blockDim.x + threadIdx.x;
  if (idx >= K * Nn) return;
  int k = idx / Nn, n = idx % Nn;
  wt[(size_t)n * K + k] = f2bf(w[idx]);
}

// -------- f32 -> bf16 activation convert --------
__global__ void a2bf_kernel(const float* __restrict__ a, __bf16* __restrict__ o, int n) {
  int idx = blockIdx.x * blockDim.x + threadIdx.x;
  if (idx < n) o[idx] = f2bf(a[idx]);
}

// -------- WMMA bf16 GEMM: C(M,Nn) = A(M,K) * Bt(Nn,K)^T + bias --------
// One wave per 16x16 tile; grids sized so every wave is full (EXEC all-1s).
__global__ void gemm_bf16_wmma(const __bf16* __restrict__ A, const __bf16* __restrict__ Bt,
                               const float* __restrict__ bias, float* __restrict__ C,
                               int M, int Nn, int K) {
  int gid    = blockIdx.x * blockDim.x + threadIdx.x;
  int wid    = gid >> 5;
  int lane   = threadIdx.x & 31;
  int ntiles = Nn >> 4;
  int mt = wid / ntiles;
  int nt = wid - mt * ntiles;
  int m  = lane & 15;
  int hh = lane >> 4;
  const __bf16* Arow = A  + ((size_t)(mt * 16 + m)) * K;
  const __bf16* Brow = Bt + ((size_t)(nt * 16 + m)) * K;   // column n = lane&15
  v8f acc = {};
  for (int k0 = 0; k0 < K; k0 += 32) {
    BfFrag a, bm;
    // A 16x32: lane holds K = {8h+e : e<8} U {16+8h+e-8 : e>=8}  (two 16B runs)
    a.p.lo = *(const v8bf*)(Arow + k0 + hh * 8);
    a.p.hi = *(const v8bf*)(Arow + k0 + 16 + hh * 8);
    // B 32x16: lane holds K = {8h+e : e<8} U {16+8h+e-8 : e>=8}  (same as A)
    bm.p.lo = *(const v8bf*)(Brow + k0 + hh * 8);
    bm.p.hi = *(const v8bf*)(Brow + k0 + 16 + hh * 8);
    acc = __builtin_amdgcn_wmma_f32_16x16x32_bf16(false, a.v, false, bm.v,
                                                  (short)0, acc, false, false);
  }
  int n = nt * 16 + m;
  float bv = bias ? bias[n] : 0.0f;
#pragma unroll
  for (int e = 0; e < 8; e++) {
    int row = mt * 16 + e + 8 * hh;     // C/D layout: VGPR e -> M = e + 8*(lane>>4)
    C[(size_t)row * Nn + n] = acc[e] + bv;
  }
}

// -------- pt_conv edge/attention: one block per point, thread = channel --------
template <int FO>
__global__ void ptconv_edge(const float* __restrict__ pos, const int* __restrict__ nbr,
                            const float* __restrict__ xlin, const float* __restrict__ asrc,
                            const float* __restrict__ adst, const float* __restrict__ posw,
                            const float* __restrict__ posb, float* __restrict__ out) {
  int point = blockIdx.x;
  int b = point / NN;
  int c = threadIdx.x;
  __shared__ int   s_nbr[KNB + 1];
  __shared__ float s_npos[(KNB + 1) * 3];
  if (c < KNB + 1) s_nbr[c] = nbr[(size_t)point * (KNB + 1) + c];
  __syncthreads();
  if (c < (KNB + 1) * 3) {
    int jj = c / 3, d = c % 3;
    s_npos[c] = pos[((size_t)b * NN + s_nbr[jj]) * 3 + d];
  }
  __syncthreads();
  float px = s_npos[0], py = s_npos[1], pz = s_npos[2];   // nbr[0] is self
  float w0 = posw[c], w1 = posw[FO + c], w2 = posw[2 * FO + c], pb = posb[c];
  float ad = adst[(size_t)point * FO + c];
  float mx = -3.0e38f, den = 0.0f, acc = 0.0f;
#pragma unroll 1
  for (int jj = 0; jj < KNB + 1; jj++) {
    int j = s_nbr[jj];
    float dx = px - s_npos[jj * 3 + 0];
    float dy = py - s_npos[jj * 3 + 1];
    float dz = pz - s_npos[jj * 3 + 2];
    float delta = fmaf(w0, dx, fmaf(w1, dy, fmaf(w2, dz, pb)));
    size_t off = ((size_t)b * NN + j) * FO + c;
    float logit = ad - asrc[off] + delta;
    float val   = xlin[off] + delta;
    float mn = fmaxf(mx, logit);        // online softmax over 17 neighbors
    float sc = __expf(mx - mn);
    float e  = __expf(logit - mn);
    den = den * sc + e;
    acc = acc * sc + e * val;
    mx = mn;
  }
  out[(size_t)point * FO + c] = acc / den;
}

// -------- final max over N --------
__global__ void maxreduce_kernel(const float* __restrict__ h, float* __restrict__ out) {
  int b = blockIdx.x, c = threadIdx.x;
  float m = -3.0e38f;
  const float* hb = h + (size_t)b * NN * LAT + c;
  for (int n = 0; n < NN; n++) m = fmaxf(m, hb[(size_t)n * LAT]);
  out[b * LAT + c] = m;
}

extern "C" void kernel_launch(void* const* d_in, const int* in_sizes, int n_in,
                              void* d_out, int out_size, void* d_ws, size_t ws_size,
                              hipStream_t stream) {
  (void)in_sizes; (void)n_in; (void)out_size; (void)ws_size;
  const float* x     = (const float*)d_in[0];
  const float* w_in1 = (const float*)d_in[1];
  const float* b_in1 = (const float*)d_in[2];
  const float* w_in2 = (const float*)d_in[3];
  const float* b_in2 = (const float*)d_in[4];
  const float* ptw[3][5];
  for (int l = 0; l < 3; l++)
    for (int q = 0; q < 5; q++)
      ptw[l][q] = (const float*)d_in[5 + l * 5 + q];

  // bump allocator over workspace
  char* ws = (char*)d_ws;
  auto alloc = [&](size_t bytes) -> void* {
    void* p = (void*)ws;
    ws += (bytes + 255) & ~(size_t)255;
    return p;
  };
  float*  pos  = (float*) alloc((size_t)BN * 3 * sizeof(float));
  int*    nbr  = (int*)   alloc((size_t)BN * (KNB + 1) * sizeof(int));
  __bf16* hbf  = (__bf16*)alloc((size_t)BN * HID * sizeof(__bf16));
  float*  h0   = (float*) alloc((size_t)BN * HID * sizeof(float));
  float*  h1   = (float*) alloc((size_t)BN * HID * sizeof(float));
  float*  h3   = (float*) alloc((size_t)BN * LAT * sizeof(float));
  float*  xlin = (float*) alloc((size_t)BN * LAT * sizeof(float));
  float*  asrc = (float*) alloc((size_t)BN * LAT * sizeof(float));
  float*  adst = (float*) alloc((size_t)BN * LAT * sizeof(float));
  __bf16* wt_in2 = (__bf16*)alloc((size_t)HID * HID * sizeof(__bf16));
  __bf16* wt[3][3];
  for (int l = 0; l < 3; l++)
    for (int q = 0; q < 3; q++)
      wt[l][q] = (__bf16*)alloc((size_t)HID * LAT * sizeof(__bf16));

  // geometry
  prep_pos<<<BN / 256, 256, 0, stream>>>(x, pos);
  knn_kernel<<<dim3(NN / 256, BB), 256, 0, stream>>>(pos, nbr);

  // weights -> transposed bf16
  wcvt_kernel<<<(HID * HID + 255) / 256, 256, 0, stream>>>(w_in2, wt_in2, HID, HID);
  int fo_l[3] = {HID, HID, LAT};
  for (int l = 0; l < 3; l++)
    for (int q = 0; q < 3; q++)
      wcvt_kernel<<<(HID * fo_l[l] + 255) / 256, 256, 0, stream>>>(ptw[l][q], wt[l][q],
                                                                   HID, fo_l[l]);

  // input MLP: relu(xt@w1+b1) -> bf16, then WMMA GEMM with w_in2 (+b_in2)
  mlp1_kernel<<<(BN * HID) / 256, 256, 0, stream>>>(x, w_in1, b_in1, hbf);
  int tiles0 = (BN / 16) * (HID / 16);
  gemm_bf16_wmma<<<tiles0 / 8, 256, 0, stream>>>(hbf, wt_in2, b_in2, h0, BN, HID, HID);

  // three pt_conv layers
  float* hin = h0;
  float* hout128[2] = {h1, h0};
  for (int l = 0; l < 3; l++) {
    int fo = fo_l[l];
    a2bf_kernel<<<(BN * HID) / 256, 256, 0, stream>>>(hin, hbf, BN * HID);
    int tiles = (BN / 16) * (fo / 16);
    gemm_bf16_wmma<<<tiles / 8, 256, 0, stream>>>(hbf, wt[l][0], nullptr, xlin, BN, fo, HID);
    gemm_bf16_wmma<<<tiles / 8, 256, 0, stream>>>(hbf, wt[l][1], nullptr, asrc, BN, fo, HID);
    gemm_bf16_wmma<<<tiles / 8, 256, 0, stream>>>(hbf, wt[l][2], nullptr, adst, BN, fo, HID);
    float* hdst = (l == 2) ? h3 : hout128[l];
    if (fo == HID)
      ptconv_edge<HID><<<BN, HID, 0, stream>>>(pos, nbr, xlin, asrc, adst,
                                               ptw[l][3], ptw[l][4], hdst);
    else
      ptconv_edge<LAT><<<BN, LAT, 0, stream>>>(pos, nbr, xlin, asrc, adst,
                                               ptw[l][3], ptw[l][4], hdst);
    hin = hdst;
  }

  maxreduce_kernel<<<BB, LAT, 0, stream>>>(h3, (float*)d_out);
}
